// MultiHeadAttention_83743272337487
// MI455X (gfx1250) — matrix operand order repaired and checked by reading, not compiled, not run on hardware
//
#include <hip/hip_runtime.h>
#include <hip/hip_bf16.h>

// ---------------- problem constants ----------------
#define B_  2
#define L_  4096
#define D_  768
#define H_  12
#define HS_ 64

// ---------------- types ----------------
typedef __attribute__((ext_vector_type(16))) __bf16       v16bf;
typedef __attribute__((ext_vector_type(8)))  float        v8f;
typedef __attribute__((ext_vector_type(4)))  unsigned int u32x4;
typedef __attribute__((ext_vector_type(8)))  int          i32x8;
typedef __attribute__((ext_vector_type(4)))  int          i32x4;

struct Frag32 { u32x4 lo, hi; };  // 32 bytes == v16bf

static __device__ __forceinline__ v16bf as_v16bf(Frag32 f) {
  return __builtin_bit_cast(v16bf, f);
}

static __device__ __forceinline__ v8f vzero8() {
  v8f z;
#pragma unroll
  for (int i = 0; i < 8; ++i) z[i] = 0.0f;
  return z;
}

static __device__ __forceinline__ unsigned short f32_bf16(float f) {
  unsigned int u = __float_as_uint(f);
  u += 0x7FFFu + ((u >> 16) & 1u);          // round-to-nearest-even
  return (unsigned short)(u >> 16);
}

// ---------------- CDNA5 async global->LDS copy (ASYNCcnt path) ----------------
static __device__ __forceinline__ void async_copy_b128(unsigned lds_byte_off, const void* gptr) {
  unsigned long long ga = (unsigned long long)gptr;
  asm volatile("global_load_async_to_lds_b128 %0, %1, off"
               :: "v"(lds_byte_off), "v"(ga) : "memory");
}
static __device__ __forceinline__ void wait_async0() {
#if __has_builtin(__builtin_amdgcn_s_wait_asynccnt)
  __builtin_amdgcn_s_wait_asynccnt(0);
#else
  asm volatile("s_wait_asynccnt 0x0" ::: "memory");
#endif
}
static __device__ __forceinline__ void wait_tensor0() {
#if __has_builtin(__builtin_amdgcn_s_wait_tensorcnt)
  __builtin_amdgcn_s_wait_tensorcnt((short)0);
#else
  asm volatile("s_wait_tensorcnt 0x0" ::: "memory");
#endif
}
static __device__ __forceinline__ unsigned lds_off(const void* p) {
  return (unsigned)(unsigned long long)p;   // LDS aperture: addr[31:0] is the LDS offset
}

// ---------------- CDNA5 Tensor Data Mover staging (TENSORcnt path) ----------------
// Stages a 64-row x 32-element bf16 tile (tensor row stride D_ elements) into LDS with
// 16B padding after each 64B row (-> LDS row stride WSTR=40 elements).
#define WSTR 40
#if __has_builtin(__builtin_amdgcn_tensor_load_to_lds)
#define HAVE_TDM 1
#else
#define HAVE_TDM 0
#endif

static __device__ __forceinline__ void tdm_stage_tile64x32(unsigned lds_byte, const void* g) {
#if HAVE_TDM
  unsigned long long ga = (unsigned long long)g;
  u32x4 g0;
  g0[0] = 1u;                                   // count=1 (valid user descriptor)
  g0[1] = lds_byte;                             // LDS byte address
  g0[2] = (unsigned)ga;                         // global_addr[31:0]
  g0[3] = ((unsigned)(ga >> 32) & 0x01FFFFFFu)  // global_addr[56:32]
        | (2u << 30);                           // type=2 ("image")
  i32x8 g1;
  g1[0] = (1 << 16)                             // data_size = 2B
        | (1 << 20)                             // pad_enable
        | (3 << 22)                             // pad_interval: every 16 DW (64B)
        | (3 << 25);                            // pad_amount: 4 DW (16B)
  g1[1] = 0;                                    // atomic_barrier off; tensor_dim0 lo16 (1<<20 -> 0)
  g1[2] = 16;                                   // tensor_dim0 hi16 = 16 ; tensor_dim1 lo16 = 0
  g1[3] = 16 | (32 << 16);                      // tensor_dim1 hi16 = 16 ; tile_dim0 = 32
  g1[4] = 64;                                   // tile_dim1 = 64 ; tile_dim2 = 0
  g1[5] = D_;                                   // tensor_dim0_stride = 768 (lo32)
  g1[6] = 0;                                    // stride hi16 ; dim1_stride lo16
  g1[7] = 0;
  i32x4 z4 = {0, 0, 0, 0};
#if defined(__clang_major__) && (__clang_major__ >= 23)
  i32x8 z8 = {0, 0, 0, 0, 0, 0, 0, 0};
  __builtin_amdgcn_tensor_load_to_lds(g0, g1, z4, z4, z8, 0);
#else
  __builtin_amdgcn_tensor_load_to_lds(g0, g1, z4, z4, 0);
#endif
#else
  (void)lds_byte; (void)g;
#endif
}

// Stage one weight chunk (64 cols x 32 d) into LDS: TDM if available, else per-lane async.
static __device__ __forceinline__ void stage_w_chunk(unsigned short* lds, const unsigned short* g,
                                                     int tid, int w) {
#if HAVE_TDM
  (void)tid;
  if (w == 0) tdm_stage_tile64x32(lds_off(lds), g);
#else
  (void)w;
  int scol = tid >> 1;                 // 0..63
  int soff = (tid & 1) * 16;           // 0 or 16 (elements)
#pragma unroll
  for (int i = 0; i < 2; ++i)
    async_copy_b128(lds_off(lds + scol * WSTR + soff + i * 8),
                    g + (size_t)scol * D_ + soff + i * 8);
#endif
}
static __device__ __forceinline__ void stage_w_wait(int w) {
#if HAVE_TDM
  if (w == 0) wait_tensor0();
#else
  (void)w;
  wait_async0();
#endif
}

// ---------------- WMMA fragment loaders ----------------
// A-matrix 16x32 bf16 layout (ISA 7.12.2):
//   lane m (m=lane&15), half=lane>>4 : K = half*8 + {0..7} and half*8+16 + {0..7}
static __device__ __forceinline__ v16bf load_A(const unsigned short* base, int ld, int lane) {
  int m = lane & 15, half = lane >> 4;
  const unsigned short* p = base + (size_t)m * ld + half * 8;
  Frag32 f;
  f.lo = *(const u32x4*)(p);
  f.hi = *(const u32x4*)(p + 16);
  return as_v16bf(f);
}

// B-matrix 32x16 bf16 layout: lane n (n=lane&15) holds column n;
// half=lane>>4 : K = half*8 + {0..7} and half*8+16 + {0..7} (same as A).
static __device__ __forceinline__ v16bf load_B(const unsigned short* base, int ld, int lane) {
  int n = lane & 15, half = lane >> 4;
  const unsigned short* p = base + (size_t)n * ld + half * 8;
  Frag32 f;
  f.lo = *(const u32x4*)(p);
  f.hi = *(const u32x4*)(p + 16);
  return as_v16bf(f);
}

static __device__ __forceinline__ v8f wmma_bf16(v16bf a, v16bf b, v8f c) {
  return __builtin_amdgcn_wmma_f32_16x16x32_bf16(false, a, false, b, (short)0, c, false, false);
}

// ---------------- kernel 0: f32 -> bf16 ----------------
__global__ void mha_cvt_bf16(const float* __restrict__ src,
                             unsigned short* __restrict__ dst, int n) {
  int i = blockIdx.x * blockDim.x + threadIdx.x;
  int stride = gridDim.x * blockDim.x;
  for (; i < n; i += stride) dst[i] = f32_bf16(src[i]);
}

// ---------------- kernel 1: Q projection  q[b,h,l,e] = sum_d x[b,l,d]*Wq[h,e,d] ----------------
// grid: (L/64, B*H), block 128 (4 waves). Shared weight tile double-buffered in LDS,
// staged by the Tensor Data Mover (wave 0), overlapped with WMMA on the current tile.
__global__ void __launch_bounds__(128) mha_qproj(const unsigned short* __restrict__ xb,
                                                 const unsigned short* __restrict__ wqb,
                                                 unsigned short* __restrict__ qb) {
  __shared__ unsigned short Wl[2][64 * WSTR];

  int bh = blockIdx.y;
  int b = bh / H_, h = bh % H_;
  int tid = threadIdx.x;
  int w = tid >> 5, lane = tid & 31;
  int m = lane & 15, half = lane >> 4;
  int row0 = blockIdx.x * 64 + w * 16;

  const unsigned short* xrow = xb + ((size_t)b * L_ + row0) * D_;
  const unsigned short* wh   = wqb + (size_t)h * HS_ * D_;

  stage_w_chunk(&Wl[0][0], wh, tid, w);
  stage_w_wait(w);
  __syncthreads();

  v8f acc[4];
#pragma unroll
  for (int t = 0; t < 4; ++t) acc[t] = vzero8();

  const int NIT = D_ / 32;   // 24
  for (int it = 0; it < NIT; ++it) {
    int cur = it & 1, nxt = cur ^ 1;
    int d0 = it * 32;
    if (it + 1 < NIT) stage_w_chunk(&Wl[nxt][0], wh + (d0 + 32), tid, w);
    if (d0 + 64 < D_) __builtin_prefetch(xrow + d0 + 64, 0, 3);
    v16bf a = load_A(xrow + d0, D_, lane);
#pragma unroll
    for (int t = 0; t < 4; ++t) {
      v16bf bm = load_B(&Wl[cur][(t * 16) * WSTR], WSTR, lane);
      acc[t] = wmma_bf16(a, bm, acc[t]);
    }
    stage_w_wait(w);
    __syncthreads();
  }

  unsigned short* qhead = qb + ((size_t)bh * L_ + row0) * HS_;
#pragma unroll
  for (int t = 0; t < 4; ++t)
#pragma unroll
    for (int r = 0; r < 8; ++r) {
      int M = r + half * 8;
      qhead[(size_t)M * HS_ + t * 16 + m] = f32_bf16(acc[t][r]);
    }
}

// ---------------- kernel 2: causal flash attention (q == k == v) ----------------
// grid: (L/64, B*H), block 128 (4 waves). K tile staged via async-to-LDS (ASYNCcnt);
// transposed V tile built by an LDS->LDS pass; online softmax in exp2 domain.
__global__ void __launch_bounds__(128) mha_flash(const unsigned short* __restrict__ qb,
                                                 unsigned short* __restrict__ attnb) {
  __shared__ unsigned short Kl[64 * 72];      // [key][e]  (padded stride 72)
  __shared__ unsigned short Vt[64 * 72];      // [e][key]  (padded stride 72)
  __shared__ unsigned short Pbuf[4 * 16 * 64];// per-wave P staging

  int bh = blockIdx.y;
  int b = bh / H_, h = bh % H_;
  int tid = threadIdx.x;
  int w = tid >> 5, lane = tid & 31;
  int m = lane & 15, half = lane >> 4;
  int qblk = blockIdx.x;
  int qrow0 = qblk * 64 + w * 16;

  const unsigned short* qhead = qb + (size_t)bh * L_ * HS_;

  v16bf qA0 = load_A(qhead + (size_t)qrow0 * HS_,      HS_, lane);
  v16bf qA1 = load_A(qhead + (size_t)qrow0 * HS_ + 32, HS_, lane);

  v8f O[4];
#pragma unroll
  for (int t = 0; t < 4; ++t) O[t] = vzero8();
  float mrun[8], lrun[8];
#pragma unroll
  for (int r = 0; r < 8; ++r) { mrun[r] = -1e30f; lrun[r] = 0.0f; }

  unsigned short* Pw = Pbuf + w * (16 * 64);
  const float kScale = 0.036084391824351614f * 1.4426950408889634f; // 1/sqrt(768)*log2(e)

  for (int kb = 0; kb <= qblk; ++kb) {
    int key0 = kb * 64;
    __syncthreads();   // previous tile fully consumed before overwrite

    // async stage: 64 keys x 128B = 512 x 16B chunks over 128 threads
#pragma unroll
    for (int i = 0; i < 4; ++i) {
      int cid = tid + i * 128;
      int key = cid >> 3;
      int off = (cid & 7) * 8;
      async_copy_b128(lds_off(&Kl[key * 72 + off]),
                      qhead + (size_t)(key0 + key) * HS_ + off);
    }
    wait_async0();
    __syncthreads();

    // LDS->LDS transpose pass: Vt[e][key] = Kl[key][e]
#pragma unroll
    for (int i = 0; i < 4; ++i) {
      int cid = tid + i * 128;
      int key = cid >> 3;
      int off = (cid & 7) * 8;
      u32x4 d = *(const u32x4*)(&Kl[key * 72 + off]);
#pragma unroll
      for (int j = 0; j < 4; ++j) {
        unsigned int word = d[j];
        Vt[(off + 2 * j)     * 72 + key] = (unsigned short)(word & 0xFFFFu);
        Vt[(off + 2 * j + 1) * 72 + key] = (unsigned short)(word >> 16);
      }
    }
    __syncthreads();

    // scores: 4 tiles of 16x16, contraction over e=64
    v8f st[4];
#pragma unroll
    for (int t = 0; t < 4; ++t) {
      v16bf b0 = load_B(&Kl[(t * 16) * 72],      72, lane);
      v16bf b1 = load_B(&Kl[(t * 16) * 72 + 32], 72, lane);
      v8f s = vzero8();
      s = wmma_bf16(qA0, b0, s);
      s = wmma_bf16(qA1, b1, s);
      st[t] = s;
    }

    bool diag = (kb == qblk);
#pragma unroll
    for (int t = 0; t < 4; ++t)
#pragma unroll
      for (int r = 0; r < 8; ++r) {
        float v = st[t][r] * kScale;
        if (diag) {
          int key = key0 + t * 16 + m;
          int qr  = qrow0 + r + half * 8;
          if (key > qr) v = -1e30f;
        }
        st[t][r] = v;
      }

    // online softmax: rows live in VGPR index, columns across 16-lane groups
#pragma unroll
    for (int r = 0; r < 8; ++r) {
      float tm = fmaxf(fmaxf(st[0][r], st[1][r]), fmaxf(st[2][r], st[3][r]));
      tm = fmaxf(tm, __shfl_xor(tm, 1, 16));
      tm = fmaxf(tm, __shfl_xor(tm, 2, 16));
      tm = fmaxf(tm, __shfl_xor(tm, 4, 16));
      tm = fmaxf(tm, __shfl_xor(tm, 8, 16));
      float nm = fmaxf(mrun[r], tm);
      float corr = exp2f(mrun[r] - nm);
      mrun[r] = nm;
      float rs = 0.0f;
#pragma unroll
      for (int t = 0; t < 4; ++t) {
        float p = exp2f(st[t][r] - nm);
        st[t][r] = p;
        rs += p;
      }
      rs += __shfl_xor(rs, 1, 16);
      rs += __shfl_xor(rs, 2, 16);
      rs += __shfl_xor(rs, 4, 16);
      rs += __shfl_xor(rs, 8, 16);
      lrun[r] = lrun[r] * corr + rs;
#pragma unroll
      for (int t = 0; t < 4; ++t) O[t][r] *= corr;
    }

    // P: C-layout -> bf16 row-major in per-wave LDS -> reload in A layout
#pragma unroll
    for (int t = 0; t < 4; ++t)
#pragma unroll
      for (int r = 0; r < 8; ++r)
        Pw[(size_t)(r + half * 8) * 64 + t * 16 + m] = f32_bf16(st[t][r]);

    v16bf pA0 = load_A(Pw,      64, lane);   // keys [0,32)
    v16bf pA1 = load_A(Pw + 32, 64, lane);   // keys [32,64)

    // O += P @ V : B columns are rows of Vt (contiguous over key)
#pragma unroll
    for (int t = 0; t < 4; ++t) {
      v16bf vb0 = load_B(&Vt[(t * 16) * 72],      72, lane);
      v16bf vb1 = load_B(&Vt[(t * 16) * 72 + 32], 72, lane);
      O[t] = wmma_bf16(pA0, vb0, O[t]);
      O[t] = wmma_bf16(pA1, vb1, O[t]);
    }
  }

  // normalize and write concat-head layout attn[b, l, h*64 + e] (bf16)
#pragma unroll
  for (int t = 0; t < 4; ++t)
#pragma unroll
    for (int r = 0; r < 8; ++r) {
      int M = r + half * 8;
      float v = O[t][r] / lrun[r];
      size_t idx = ((size_t)b * L_ + (qrow0 + M)) * D_ + h * 64 + t * 16 + m;
      attnb[idx] = f32_bf16(v);
    }
}

// ---------------- kernel 3: output projection  out = attn @ W_out^T + b_out ----------------
// grid: (B*L/64, D/64), block 128 (4 waves). TDM-staged double-buffered weights.
__global__ void __launch_bounds__(128) mha_oproj(const unsigned short* __restrict__ attnb,
                                                 const unsigned short* __restrict__ wob,
                                                 const float* __restrict__ bias,
                                                 float* __restrict__ out) {
  __shared__ unsigned short Wl[2][64 * WSTR];

  int tid = threadIdx.x;
  int w = tid >> 5, lane = tid & 31;
  int m = lane & 15, half = lane >> 4;
  int row0 = blockIdx.x * 64 + w * 16;   // over B*L rows
  int col0 = blockIdx.y * 64;

  const unsigned short* wcol = wob + (size_t)col0 * D_;
  const unsigned short* arow = attnb + (size_t)row0 * D_;

  stage_w_chunk(&Wl[0][0], wcol, tid, w);
  stage_w_wait(w);
  __syncthreads();

  v8f acc[4];
#pragma unroll
  for (int t = 0; t < 4; ++t) {
    float bv = bias[col0 + t * 16 + m];
#pragma unroll
    for (int r = 0; r < 8; ++r) acc[t][r] = bv;
  }

  const int NIT = D_ / 32;
  for (int it = 0; it < NIT; ++it) {
    int cur = it & 1, nxt = cur ^ 1;
    int d0 = it * 32;
    if (it + 1 < NIT) stage_w_chunk(&Wl[nxt][0], wcol + (d0 + 32), tid, w);
    if (d0 + 64 < D_) __builtin_prefetch(arow + d0 + 64, 0, 3);
    v16bf a = load_A(arow + d0, D_, lane);
#pragma unroll
    for (int t = 0; t < 4; ++t) {
      v16bf bm = load_B(&Wl[cur][(t * 16) * WSTR], WSTR, lane);
      acc[t] = wmma_bf16(a, bm, acc[t]);
    }
    stage_w_wait(w);
    __syncthreads();
  }

#pragma unroll
  for (int t = 0; t < 4; ++t)
#pragma unroll
    for (int r = 0; r < 8; ++r) {
      int M = r + half * 8;
      out[(size_t)(row0 + M) * D_ + col0 + t * 16 + m] = acc[t][r];
    }
}

// ---------------- host launcher ----------------
extern "C" void kernel_launch(void* const* d_in, const int* in_sizes, int n_in,
                              void* d_out, int out_size, void* d_ws, size_t ws_size,
                              hipStream_t stream) {
  (void)in_sizes; (void)n_in; (void)out_size; (void)ws_size;
  const float* x  = (const float*)d_in[0];   // [B,L,D]
  const float* Wq = (const float*)d_in[1];   // [H,HS,D]
  const float* Wo = (const float*)d_in[2];   // [D,D]
  const float* bo = (const float*)d_in[3];   // [D]

  char* ws = (char*)d_ws;
  size_t off = 0;
  auto carve = [&](size_t bytes) -> void* {
    void* p = ws + off;
    off += (bytes + 255) & ~(size_t)255;
    return p;
  };
  const size_t nX  = (size_t)B_ * L_ * D_;
  const size_t nWq = (size_t)H_ * HS_ * D_;
  const size_t nWo = (size_t)D_ * D_;
  const size_t nQ  = (size_t)B_ * H_ * L_ * HS_;

  unsigned short* xb    = (unsigned short*)carve(nX  * 2);
  unsigned short* wqb   = (unsigned short*)carve(nWq * 2);
  unsigned short* wob   = (unsigned short*)carve(nWo * 2);
  unsigned short* qbuf  = (unsigned short*)carve(nQ  * 2);
  unsigned short* attnb = (unsigned short*)carve(nX  * 2);

  mha_cvt_bf16<<<4096, 256, 0, stream>>>(x,  xb,  (int)nX);
  mha_cvt_bf16<<<1024, 256, 0, stream>>>(Wq, wqb, (int)nWq);
  mha_cvt_bf16<<<1024, 256, 0, stream>>>(Wo, wob, (int)nWo);

  dim3 gq(L_ / 64, B_ * H_);
  mha_qproj<<<gq, 128, 0, stream>>>(xb, wqb, qbuf);

  dim3 gf(L_ / 64, B_ * H_);
  mha_flash<<<gf, 128, 0, stream>>>(qbuf, attnb);

  dim3 go((B_ * L_) / 64, D_ / 64);
  mha_oproj<<<go, 128, 0, stream>>>(attnb, wob, bo, (float*)d_out);
}
